// SwinTransformerBlock3D_747324309602
// MI455X (gfx1250) — matrix operand order repaired and checked by reading, not compiled, not run on hardware
//
#include <hip/hip_runtime.h>

// ---------------- types & helpers ----------------
typedef __attribute__((ext_vector_type(16))) __bf16 v16bf;
typedef __attribute__((ext_vector_type(8)))  __bf16 v8bf;
typedef __attribute__((ext_vector_type(8)))  float  v8f;

union V16 { v16bf v; v8bf h[2]; unsigned short u[16]; };

#define WMMA_BF16(A,B,C) \
  __builtin_amdgcn_wmma_f32_16x16x32_bf16(false,(A),false,(B),(short)0,(C),false,false)

__device__ __forceinline__ unsigned short f2bf(float f) {
  unsigned int u = __float_as_uint(f);
  return (unsigned short)((u + 0x7fffu + ((u >> 16) & 1u)) >> 16);
}

// CDNA5 async global->LDS copy (ASYNCcnt path), GV addressing mode.
__device__ __forceinline__ void async_ld_b128(unsigned lds_off, const void* gaddr) {
  asm volatile("global_load_async_to_lds_b128 %0, %1, off"
               :: "v"(lds_off), "v"(gaddr)
               : "memory");
}
__device__ __forceinline__ void wait_async0() {
  asm volatile("s_wait_asynccnt 0" ::: "memory");
}

// ---------------- fp32 -> bf16 weight conversion ----------------
__global__ void k_f2bf(const float* __restrict__ s, unsigned short* __restrict__ d, int n) {
  int i = blockIdx.x * 256 + threadIdx.x;
  if (i < n) d[i] = f2bf(s[i]);
}

// ---------------- LN1 + roll(-1,-3,-3) + window partition -> bf16 ----------------
__global__ __launch_bounds__(256) void k_ln_win(
    const float* __restrict__ x, const float* __restrict__ gw,
    const float* __restrict__ gb, unsigned short* __restrict__ xw) {
  const int d = blockIdx.x;              // window-token index 0..50175
  const int wi = d / 98, n = d % 98;
  const int b = wi >> 8, wrem = wi & 255;
  const int wti = wrem >> 6, whi = (wrem >> 3) & 7, wwi = wrem & 7;
  const int it = n / 49, r = n % 49, ih = r / 7, iw = r % 7;
  const int t = wti * 2 + it, h = whi * 7 + ih, w = wwi * 7 + iw;
  const int tt = (t + 1) & 7;
  int hh = h + 3; if (hh >= 56) hh -= 56;
  int ww = w + 3; if (ww >= 56) ww -= 56;
  const int src = (((b * 8 + tt) * 56 + hh) * 56 + ww) * 256;
  const int c = threadIdx.x;
  float v = x[src + c];
  __shared__ float sb[256];
  __shared__ float s_mu, s_rv;
  sb[c] = v; __syncthreads();
  for (int s = 128; s > 0; s >>= 1) { if (c < s) sb[c] += sb[c + s]; __syncthreads(); }
  if (c == 0) s_mu = sb[0] * (1.f / 256.f);
  __syncthreads();
  float dd = v - s_mu;
  sb[c] = dd * dd; __syncthreads();
  for (int s = 128; s > 0; s >>= 1) { if (c < s) sb[c] += sb[c + s]; __syncthreads(); }
  if (c == 0) s_rv = rsqrtf(sb[0] * (1.f / 256.f) + 1e-5f);
  __syncthreads();
  xw[d * 256 + c] = f2bf(dd * s_rv * gw[c] + gb[c]);
}

// ---------------- WMMA GEMM: out = act(A @ W^T + bias [+ res]) ----------------
// A: MxK bf16 row-major, W: NxK bf16 row-major.
// 128x128 block tile, 8 waves, each wave = 64x32 register tile (4x2 WMMA accums).
// K-panels (128x32) double-buffered in LDS, filled with async global->LDS b128.
template<bool OBF, bool GELU_, bool RES>
__global__ __launch_bounds__(256) void k_gemm(
    const unsigned short* __restrict__ A, const unsigned short* __restrict__ W,
    const float* __restrict__ bias, void* __restrict__ outv,
    const float* __restrict__ res, int M, int N, int K) {
  __shared__ __align__(16) unsigned short As[2][128][32];
  __shared__ __align__(16) unsigned short Ws[2][128][32];
  const int tid = threadIdx.x;
  const int lane = tid & 31, wav = tid >> 5;
  const int lr = lane & 15, hi = lane >> 4, co = hi << 3;
  const int wm = wav >> 2, wn = wav & 3;            // 2x4 wave grid
  const int mblk = blockIdx.y * 128, nblk = blockIdx.x * 128;

  auto stage = [&](int buf, int k0) {
    // 128 rows x 32 bf16 = 512 x 16B chunks per matrix; 2 chunks/thread each
    for (int e = tid; e < 512; e += 256) {
      const int row = e >> 2, q = e & 3;
      const char* ga = (const char*)A + ((size_t)(mblk + row) * K + k0) * 2 + q * 16;
      const char* gw = (const char*)W + ((size_t)(nblk + row) * K + k0) * 2 + q * 16;
      async_ld_b128((unsigned)(size_t)&As[buf][row][q * 8], ga);
      async_ld_b128((unsigned)(size_t)&Ws[buf][row][q * 8], gw);
    }
  };

  v8f acc[4][2];
  for (int i = 0; i < 4; ++i)
    for (int j = 0; j < 2; ++j)
      acc[i][j] = (v8f){0.f,0.f,0.f,0.f,0.f,0.f,0.f,0.f};

  stage(0, 0);
  wait_async0();
  __syncthreads();

  int buf = 0;
  for (int k0 = 0; k0 < K; k0 += 32) {
    if (k0 + 32 < K) stage(buf ^ 1, k0 + 32);   // overlap fill with compute
    V16 af[4], bw[2];
    for (int i = 0; i < 4; ++i) {
      const unsigned short* p = &As[buf][wm * 64 + i * 16 + lr][0];
      af[i].h[0] = *(const v8bf*)(p + co);
      af[i].h[1] = *(const v8bf*)(p + 16 + co);
    }
    for (int j = 0; j < 2; ++j) {
      const unsigned short* p = &Ws[buf][wn * 32 + j * 16 + lr][0];
      bw[j].h[0] = *(const v8bf*)(p + co);
      bw[j].h[1] = *(const v8bf*)(p + 16 + co);
    }
    for (int i = 0; i < 4; ++i)
      for (int j = 0; j < 2; ++j)
        acc[i][j] = WMMA_BF16(af[i].v, bw[j].v, acc[i][j]);
    wait_async0();
    __syncthreads();
    buf ^= 1;
  }

  for (int j = 0; j < 2; ++j) {
    const int n = nblk + wn * 32 + j * 16 + lr;
    const float bn = bias[n];
    for (int i = 0; i < 4; ++i) {
      for (int p = 0; p < 8; ++p) {
        const int m = mblk + wm * 64 + i * 16 + p + (hi << 3);
        float v = acc[i][j][p] + bn;
        if (GELU_) v = 0.5f * v * (1.f + erff(v * 0.70710678118f));
        if (RES)   v += res[m * N + n];
        const int o = m * N + n;
        if (OBF) ((unsigned short*)outv)[o] = f2bf(v);
        else     ((float*)outv)[o] = v;
      }
    }
  }
}

// ---------------- windowed attention: one block per (window, head) ----------------
__global__ __launch_bounds__(256) void k_attn(
    const unsigned short* __restrict__ qkv, const float* __restrict__ rpe,
    unsigned short* __restrict__ outp) {
  const int head = blockIdx.x;
  const int wi = blockIdx.y;
  const int wrem = wi & 255;
  const int wti = wrem >> 6, whi = (wrem >> 3) & 7, wwi = wrem & 7;
  const int wbase = wi * 98;

  __shared__ __align__(16) unsigned short Qs[112 * 32];
  __shared__ __align__(16) unsigned short Ks[112 * 32];
  __shared__ __align__(16) unsigned short Vt[32 * 128];   // V transposed: [ch][token]
  __shared__ __align__(16) unsigned short Ps[112 * 128];
  __shared__ int regid[112];

  const int tid = threadIdx.x;
  // async-stage Q,K rows (98 rows x 64B each = 4 chunks/row)
  const char* gq = (const char*)qkv + ((size_t)wbase * 768 + head * 32) * 2;
  for (int e = tid; e < 392; e += 256) {
    const int n = e >> 2, c4 = e & 3;
    const char* g = gq + (size_t)n * 1536 + c4 * 16;
    async_ld_b128((unsigned)(size_t)&Qs[n * 32 + c4 * 8], g);          // Q
    async_ld_b128((unsigned)(size_t)&Ks[n * 32 + c4 * 8], g + 512);    // K (+256 ch)
  }
  // V loaded transposed (regular loads; transpose can't ride the async path)
  for (int e = tid; e < 3136; e += 256) {
    const int n = e >> 5, ch = e & 31;
    Vt[ch * 128 + n] = qkv[(size_t)(wbase + n) * 768 + head * 32 + 512 + ch];
  }
  // zero pads: Q/K rows 98..111, Vt tokens 98..127
  for (int e = tid; e < 448; e += 256) {
    const int r = 98 + (e >> 5), c = e & 31;
    Qs[r * 32 + c] = 0; Ks[r * 32 + c] = 0;
  }
  for (int e = tid; e < 1024; e += 256) {
    const int k = 96 + (e >> 5), ch = e & 31;
    if (k >= 98) Vt[ch * 128 + k] = 0;
  }
  if (tid < 112) {
    int rid = 0;
    if (tid < 98) {
      const int it = tid / 49, r = tid % 49, ih = r / 7, iw = r % 7;
      const int t = wti * 2 + it, hh = whi * 7 + ih, ww = wwi * 7 + iw;
      const int rt = (t  < 6 ) ? 0 : ((t  < 7 ) ? 1 : 2);
      const int rh = (hh < 49) ? 0 : ((hh < 53) ? 1 : 2);
      const int rw = (ww < 49) ? 0 : ((ww < 53) ? 1 : 2);
      rid = rt * 9 + rh * 3 + rw;
    }
    regid[tid] = rid;
  }
  wait_async0();
  __syncthreads();

  const int lane = tid & 31, wav = tid >> 5;
  const int lr = lane & 15, hi = lane >> 4, co = hi << 3;

  // Phase 1: S = scale*Q.K^T + bias + mask; register-resident softmax -> P (bf16, LDS)
  if (wav < 7) {
    v8f acc[7];
    V16 a;
    const unsigned short* pQ = Qs + (wav * 16 + lr) * 32;
    a.h[0] = *(const v8bf*)(pQ + co);
    a.h[1] = *(const v8bf*)(pQ + 16 + co);
    for (int t = 0; t < 7; ++t) {
      V16 b;
      const unsigned short* pK = Ks + (t * 16 + lr) * 32 + co;
      b.h[0] = *(const v8bf*)(pK);
      b.h[1] = *(const v8bf*)(pK + 16);
      v8f z = {0.f,0.f,0.f,0.f,0.f,0.f,0.f,0.f};
      acc[t] = WMMA_BF16(a.v, b.v, z);
    }
    const float scale = 0.17677669529663687f;   // 32^-0.5
    for (int t = 0; t < 7; ++t) {
      const int ncol = t * 16 + lr;
      const int kid = (ncol < 98) ? regid[ncol] : 0;
      const int kt = ncol / 49, kr = ncol % 49, kh = kr / 7, kw = kr % 7;
      for (int p = 0; p < 8; ++p) {
        const int mrow = wav * 16 + p + (hi << 3);
        float s = acc[t][p] * scale;
        if (mrow < 98 && ncol < 98) {
          const int qt = mrow / 49, qr = mrow % 49, qh = qr / 7, qw = qr % 7;
          const int ridx = (qt - kt + 1) * 169 + (qh - kh + 6) * 13 + (qw - kw + 6);
          s += rpe[ridx * 8 + head];
          if (regid[mrow] != kid) s -= 100.f;
        }
        if (ncol >= 98) s = -1e30f;
        acc[t][p] = s;
      }
    }
    for (int p = 0; p < 8; ++p) {
      float mx = acc[0][p];
      for (int t = 1; t < 7; ++t) mx = fmaxf(mx, acc[t][p]);
      for (int m2 = 1; m2 < 16; m2 <<= 1) mx = fmaxf(mx, __shfl_xor(mx, m2, 32));
      float sum = 0.f;
      for (int t = 0; t < 7; ++t) { float e = __expf(acc[t][p] - mx); acc[t][p] = e; sum += e; }
      for (int m2 = 1; m2 < 16; m2 <<= 1) sum += __shfl_xor(sum, m2, 32);
      const float inv = 1.f / sum;
      unsigned short* pr = Ps + (wav * 16 + p + (hi << 3)) * 128;
      for (int t = 0; t < 7; ++t) pr[t * 16 + lr] = f2bf(acc[t][p] * inv);
      pr[112 + lr] = 0;   // zero K-pad cols 112..127
    }
  }
  __syncthreads();

  // Phase 2: O = P.V  (112x128 @ 128x32, valid rows 0..97); B frags from Vt (contiguous)
  for (int tile = wav; tile < 14; tile += 8) {
    const int mt = tile >> 1, nt = tile & 1;
    v8f acc = {0.f,0.f,0.f,0.f,0.f,0.f,0.f,0.f};
    const unsigned short* pP = Ps + (mt * 16 + lr) * 128;
    const unsigned short* pV = Vt + ((nt << 4) + lr) * 128;
    for (int k0 = 0; k0 < 128; k0 += 32) {
      V16 a, b;
      a.h[0] = *(const v8bf*)(pP + k0 + co);
      a.h[1] = *(const v8bf*)(pP + k0 + 16 + co);
      b.h[0] = *(const v8bf*)(pV + k0 + co);
      b.h[1] = *(const v8bf*)(pV + k0 + 16 + co);
      acc = WMMA_BF16(a.v, b.v, acc);
    }
    for (int p = 0; p < 8; ++p) {
      const int mrow = mt * 16 + p + (hi << 3);
      if (mrow < 98)
        outp[(wbase + mrow) * 256 + head * 32 + (nt << 4) + lr] = f2bf(acc[p]);
    }
  }
}

// ---------------- window reverse + roll(+1,+3,+3) + residual + LN2 ----------------
__global__ __launch_bounds__(256) void k_unwin_ln2(
    const float* __restrict__ x, const float* __restrict__ yproj,
    const float* __restrict__ gw, const float* __restrict__ gb,
    float* __restrict__ x2, unsigned short* __restrict__ h2) {
  const int p = blockIdx.x;            // natural token index
  const int b = p / 25088;
  int rem = p % 25088;
  const int t = rem / 3136; rem %= 3136;
  const int h = rem / 56, w = rem % 56;
  const int tt = (t + 7) & 7;
  int hh = h - 3; if (hh < 0) hh += 56;
  int ww = w - 3; if (ww < 0) ww += 56;
  const int wi = b * 256 + (tt >> 1) * 64 + (hh / 7) * 8 + (ww / 7);
  const int n = (tt & 1) * 49 + (hh % 7) * 7 + (ww % 7);
  const int c = threadIdx.x;
  float v = x[p * 256 + c] + yproj[(wi * 98 + n) * 256 + c];
  x2[p * 256 + c] = v;
  __shared__ float sb[256];
  __shared__ float s_mu, s_rv;
  sb[c] = v; __syncthreads();
  for (int s = 128; s > 0; s >>= 1) { if (c < s) sb[c] += sb[c + s]; __syncthreads(); }
  if (c == 0) s_mu = sb[0] * (1.f / 256.f);
  __syncthreads();
  float dd = v - s_mu;
  sb[c] = dd * dd; __syncthreads();
  for (int s = 128; s > 0; s >>= 1) { if (c < s) sb[c] += sb[c + s]; __syncthreads(); }
  if (c == 0) s_rv = rsqrtf(sb[0] * (1.f / 256.f) + 1e-5f);
  __syncthreads();
  h2[p * 256 + c] = f2bf(dd * s_rv * gw[c] + gb[c]);
}

// ---------------- host orchestration ----------------
extern "C" void kernel_launch(void* const* d_in, const int* in_sizes, int n_in,
                              void* d_out, int out_size, void* d_ws, size_t ws_size,
                              hipStream_t stream) {
  (void)in_sizes; (void)n_in; (void)out_size; (void)ws_size;
  const float* x      = (const float*)d_in[0];
  const float* n1w    = (const float*)d_in[1];
  const float* n1b    = (const float*)d_in[2];
  const float* qkv_w  = (const float*)d_in[3];
  const float* qkv_b  = (const float*)d_in[4];
  const float* proj_w = (const float*)d_in[5];
  const float* proj_b = (const float*)d_in[6];
  const float* rpe    = (const float*)d_in[7];
  const float* n2w    = (const float*)d_in[8];
  const float* n2b    = (const float*)d_in[9];
  const float* fc1_w  = (const float*)d_in[10];
  const float* fc1_b  = (const float*)d_in[11];
  const float* fc2_w  = (const float*)d_in[12];
  const float* fc2_b  = (const float*)d_in[13];
  float* out = (float*)d_out;
  char* ws = (char*)d_ws;
  const int M = 50176;                 // 2*8*56*56 tokens (== 512 windows * 98)

  // workspace layout (bytes). R1 slab reused: xw -> attn_out -> h2.  R2: qkv_bf16 -> y_proj.
  unsigned short* wq   = (unsigned short*)(ws + 0);          // 768x256 bf16
  unsigned short* wp   = (unsigned short*)(ws + 393216);     // 256x256 bf16
  unsigned short* w1   = (unsigned short*)(ws + 524288);     // 1024x256 bf16
  unsigned short* w2   = (unsigned short*)(ws + 1048576);    // 256x1024 bf16
  unsigned short* R1   = (unsigned short*)(ws + 1572864);    // M*256 bf16  (25.7 MB)
  unsigned short* qkvb = (unsigned short*)(ws + 27262976);   // M*768 bf16  (77 MB)
  float* yproj         = (float*)(ws + 27262976);            // f32 M*256 (aliases qkvb, used after)
  float* x2            = (float*)(ws + 104333312);           // M*256 f32
  unsigned short* h1   = (unsigned short*)(ws + 155713536);  // M*1024 bf16

  k_f2bf<<<(196608 + 255) / 256, 256, 0, stream>>>(qkv_w, wq, 196608);
  k_f2bf<<<( 65536 + 255) / 256, 256, 0, stream>>>(proj_w, wp,  65536);
  k_f2bf<<<(262144 + 255) / 256, 256, 0, stream>>>(fc1_w, w1, 262144);
  k_f2bf<<<(262144 + 255) / 256, 256, 0, stream>>>(fc2_w, w2, 262144);

  // LN1 + roll + window partition
  k_ln_win<<<M, 256, 0, stream>>>(x, n1w, n1b, R1);
  // qkv projection (bf16 out)
  k_gemm<true, false, false><<<dim3(6, 392), 256, 0, stream>>>(R1, wq, qkv_b, qkvb, nullptr, M, 768, 256);
  // windowed attention (writes attn_out into R1)
  k_attn<<<dim3(8, 512), 256, 0, stream>>>(qkvb, rpe, R1);
  // output projection (f32 out, reuses qkv slab)
  k_gemm<false, false, false><<<dim3(2, 392), 256, 0, stream>>>(R1, wp, proj_b, yproj, nullptr, M, 256, 256);
  // window reverse + roll + residual + LN2 (x2 f32, h2 bf16 into R1)
  k_unwin_ln2<<<M, 256, 0, stream>>>(x, yproj, n2w, n2b, x2, R1);
  // MLP fc1 + exact GELU (bf16 out)
  k_gemm<true, true, false><<<dim3(8, 392), 256, 0, stream>>>(R1, w1, fc1_b, h1, nullptr, M, 1024, 256);
  // MLP fc2 + residual add -> final f32 output
  k_gemm<false, false, true><<<dim3(2, 392), 256, 0, stream>>>(h1, w2, fc2_b, out, x2, M, 256, 1024);
}
